// UResNet_87660282511911
// MI455X (gfx1250) — matrix operand order repaired and checked by reading, not compiled, not run on hardware
//
#include <hip/hip_runtime.h>

// ---------------------------------------------------------------------------
// CDNA5 (gfx1250) sparse U-ResNet forward.
// Gather-GEMMs (submanifold 3x3x3, strided 8-tap, 1x1 nin) run on the WMMA
// pipe: v_wmma_f32_16x16x32_bf16 with f32 accumulate.
//  - Weights pre-packed once per launch into the exact per-lane B-fragment
//    layout (bf16, 32B-aligned) -> B fragment = one 32-byte vector load.
//  - BN+ReLU writes activations as bf16 row-major (+ zero pad row) -> A
//    fragment = 1-2 unconditional b128 loads (C is always a multiple of 16).
//  - One wave computes a full 16 x D row-strip: the A fragment is gathered
//    once and reused across D/16 WMMAs (kills redundant rulebook gathers).
// Entire working set is L2-resident (192MB); HBM traffic is negligible.
// ---------------------------------------------------------------------------

typedef __attribute__((ext_vector_type(16))) __bf16 v16bf;
typedef __attribute__((ext_vector_type(8)))  __bf16 v8bf;
typedef __attribute__((ext_vector_type(8)))  float  v8f;

#define EPS_BN 1e-4f

// ---------------- BN statistics: per-channel scale/shift ----------------
__global__ void bn_stats_kernel(const float* __restrict__ X,
                                const float* __restrict__ g,
                                const float* __restrict__ b,
                                float* __restrict__ ss, int N, int C) {
  int c = blockIdx.x;
  float s1 = 0.f, s2 = 0.f;
  for (int n = threadIdx.x; n < N; n += blockDim.x) {
    float v = X[(size_t)n * C + c];
    s1 += v; s2 += v * v;
  }
  __shared__ float r1[256], r2[256];
  r1[threadIdx.x] = s1; r2[threadIdx.x] = s2;
  __syncthreads();
  for (int s = blockDim.x >> 1; s > 0; s >>= 1) {
    if ((int)threadIdx.x < s) {
      r1[threadIdx.x] += r1[threadIdx.x + s];
      r2[threadIdx.x] += r2[threadIdx.x + s];
    }
    __syncthreads();
  }
  if (threadIdx.x == 0) {
    float mu  = r1[0] / (float)N;
    float var = r2[0] / (float)N - mu * mu;
    float inv = rsqrtf(var + EPS_BN);
    float sc  = g[c] * inv;
    ss[c]     = sc;
    ss[C + c] = b[c] - mu * sc;
  }
}

// ---------------- BN apply + ReLU -> bf16; zeroes pad row N -------------
__global__ void bn_apply_kernel(const float* __restrict__ X,
                                const float* __restrict__ ss,
                                __bf16* __restrict__ Y, int N, int C) {
  int i = blockIdx.x * blockDim.x + threadIdx.x;
  int total = (N + 1) * C;
  if (i >= total) return;
  int n = i / C, c = i - n * C;
  float v = (n == N) ? 0.f
                     : fmaxf(0.f, X[(size_t)n * C + c] * ss[c] + ss[C + c]);
  Y[i] = (__bf16)v;
}

// ---------------- f32 -> bf16 row-major copy; zeroes pad row N ----------
__global__ void cvt_bf16_kernel(const float* __restrict__ X,
                                __bf16* __restrict__ Y, int N, int C) {
  int i = blockIdx.x * blockDim.x + threadIdx.x;
  int total = (N + 1) * C;
  if (i >= total) return;
  int n = i / C;
  Y[i] = (__bf16)((n == N) ? 0.f : X[i]);
}

// ---------------- weight repack: [K,C,D] f32 -> B-fragment bf16 ---------
// out[(((k*CB+cb)*D + d)*2 + hi)*16 + i] = W[k, cb*32+(i&8)*2+hi*8+(i&7), d]  (0 if c>=C)
__global__ void repack_w_kernel(const float* __restrict__ W,
                                __bf16* __restrict__ out,
                                int C, int D, int K, int CB) {
  int o = blockIdx.x * blockDim.x + threadIdx.x;
  int total = K * CB * D * 32;
  if (o >= total) return;
  int i   = o & 15;
  int hi  = (o >> 4) & 1;
  int rest = o >> 5;                 // (k*CB+cb)*D + d
  int d    = rest % D;
  int rest2 = rest / D;
  int cb   = rest2 % CB;
  int k    = rest2 / CB;
  int c    = cb * 32 + (i & 8) * 2 + hi * 8 + (i & 7);
  out[o] = (__bf16)((c < C) ? W[((size_t)k * C + c) * D + d] : 0.f);
}

// ---------------- strided column copy (skip save / concat-left) ---------
__global__ void copy_cols_kernel(const float* __restrict__ src,
                                 float* __restrict__ dst,
                                 int N, int W, int lds, int ldd, int colOff) {
  int i = blockIdx.x * blockDim.x + threadIdx.x;
  if (i >= N * W) return;
  int n = i / W, w = i - n * W;
  dst[(size_t)n * ldd + colOff + w] = src[(size_t)n * lds + w];
}

__global__ void zero_kernel(float* __restrict__ p, int n) {
  int i = blockIdx.x * blockDim.x + threadIdx.x;
  if (i < n) p[i] = 0.f;
}

// ---------------- gather-GEMM on the WMMA pipe (compile-time shapes) ----
// Y[n, colOff+d] (+)= sum_k sum_c X[idx(n,k), c] * W[k, c, d]
// X: bf16 [Nsrc+1, C] row-major (pad row Nsrc zeroed).
// Wp: packed bf16 B fragments. One wave32 per 16 x D output row-strip:
// the gathered A fragment is reused across the D/16 column tiles.
template <int C, int D, int K>
__global__ void subconv_wmma_kernel(const __bf16* __restrict__ X,
                                    const int*    __restrict__ nbr,
                                    const __bf16* __restrict__ Wp,
                                    float*        __restrict__ Y,
                                    int N, int Nsrc, int ldY, int colOff,
                                    int accumulate) {
  const int lane    = threadIdx.x & 31;
  const int wid     = threadIdx.x >> 5;
  const int rowTile = blockIdx.x * (blockDim.x >> 5) + wid;
  if (rowTile * 16 >= N) return;                 // uniform per wave
  const int  m     = lane & 15;
  const int  hi    = lane >> 4;
  const int  row   = rowTile * 16 + m;
  const bool rowOk = row < N;
  constexpr int CB = (C + 31) / 32;
  constexpr int DT = D / 16;                     // D % 16 == 0 always

  v8f acc[DT];
#pragma unroll
  for (int dt = 0; dt < DT; dt++) acc[dt] = {};
  if (accumulate) {
#pragma unroll
    for (int dt = 0; dt < DT; dt++)
#pragma unroll
      for (int r = 0; r < 8; r++) {
        int mr = rowTile * 16 + r + hi * 8;
        if (mr < N) acc[dt][r] = Y[(size_t)mr * ldY + colOff + dt * 16 + m];
      }
  }

  // per-lane packed-B base; +512 elems per column tile, +32*D per (k,cb)
  const __bf16* wb = Wp + ((size_t)m * 2 + hi) * 16;

  for (int k = 0; k < K; k++) {
    int src = rowOk ? (nbr ? nbr[(size_t)row * K + k] : row) : Nsrc;
    const __bf16* xr = X + (size_t)src * C;
#pragma unroll
    for (int cb = 0; cb < CB; cb++) {
      v16bf a;
      if constexpr (C == 1) {
        v16bf t = {};
        if (hi == 0) t[0] = xr[0];               // single input channel
        a = t;
      } else {
        const int cs0 = cb * 32 + hi * 8;        // always valid span
        const int cs1 = cs0 + 16;                // compile-time valid/invalid
        v8bf lo = *(const v8bf*)(xr + cs0);
        v8bf hv = {};
        if (cs1 < C) hv = *(const v8bf*)(xr + cs1);
        a = __builtin_shufflevector(lo, hv, 0, 1, 2, 3, 4, 5, 6, 7,
                                    8, 9, 10, 11, 12, 13, 14, 15);
      }
      const __bf16* wkb = wb + (size_t)(k * CB + cb) * D * 32;
#pragma unroll
      for (int dt = 0; dt < DT; dt++) {
        v16bf b = *(const v16bf*)(wkb + (size_t)dt * 512);
        acc[dt] = __builtin_amdgcn_wmma_f32_16x16x32_bf16(
            /*neg_a=*/false, a, /*neg_b=*/false, b,
            /*c_mod=*/(short)0, acc[dt], /*reuse_a=*/false, /*reuse_b=*/false);
      }
    }
  }

#pragma unroll
  for (int dt = 0; dt < DT; dt++)
#pragma unroll
    for (int r = 0; r < 8; r++) {
      int mr = rowTile * 16 + r + hi * 8;
      if (mr < N) Y[(size_t)mr * ldY + colOff + dt * 16 + m] = acc[dt][r];
    }
}

// ---------------- transpose-conv scatter (decoder upsample) -------------
__global__ void deconv_kernel(const __bf16* __restrict__ h,
                              const int* __restrict__ parent,
                              const int* __restrict__ koff,
                              const float* __restrict__ W,
                              float* __restrict__ out,
                              int Nf, int A, int B, int ldd, int colOff) {
  int i = blockIdx.x * blockDim.x + threadIdx.x;
  if (i >= Nf * B) return;
  int n = i / B, d = i - n * B;
  const __bf16* hr = h + (size_t)parent[n] * A;
  const float*  wk = W + (size_t)koff[n] * A * B;
  float acc = 0.f;
  for (int c = 0; c < A; c++) acc += (float)hr[c] * wk[(size_t)c * B + d];
  out[(size_t)n * ldd + colOff + d] = acc;
}

// ---------------- final linear head -------------------------------------
__global__ void head_kernel(const __bf16* __restrict__ h,
                            const float* __restrict__ w,
                            const float* __restrict__ bias,
                            float* __restrict__ out, int N, int C, int Q) {
  int i = blockIdx.x * blockDim.x + threadIdx.x;
  if (i >= N * Q) return;
  int n = i / Q, q = i - n * Q;
  float acc = bias[q];
  for (int c = 0; c < C; c++) acc += (float)h[(size_t)n * C + c] * w[c * Q + q];
  out[i] = acc;
}

// ===========================================================================
// Host orchestration
// ===========================================================================
struct BlkIdx { int bn1g, bn1b, bn2g, bn2b, c1, c2, nin; };
struct DecIdx { BlkIdx blk[2]; int bng, bnb, w; };
struct EncIdx { BlkIdx blk[2]; int dbng, dbnb, dw; };

struct SubArgs {
  const __bf16* X; const int* nbr; const __bf16* Wp; float* Y;
  int N, Nsrc, ldY, colOff, acc; hipStream_t s;
};

template <int C, int D, int K>
static void launch_sub(const SubArgs& a) {
  dim3 grid(((a.N + 15) / 16 + 7) / 8);
  subconv_wmma_kernel<C, D, K><<<grid, 256, 0, a.s>>>(
      a.X, a.nbr, a.Wp, a.Y, a.N, a.Nsrc, a.ldY, a.colOff, a.acc);
}

// Exhaustive over the 18 (C,D,K) triples the network uses.
static void subconv_dispatch(int C, int D, int K, const SubArgs& a) {
#define SC_CASE(c, d, k)                                   \
  if (C == (c) && D == (d) && K == (k)) { launch_sub<c, d, k>(a); return; }
  SC_CASE(1, 16, 27)                                       // stem
  SC_CASE(16, 16, 27) SC_CASE(32, 32, 27) SC_CASE(48, 48, 27)
  SC_CASE(64, 64, 27) SC_CASE(80, 80, 27)                  // enc / dec blocks
  SC_CASE(16, 32, 8) SC_CASE(32, 48, 8) SC_CASE(48, 64, 8)
  SC_CASE(64, 80, 8)                                       // down convs
  SC_CASE(32, 16, 27) SC_CASE(64, 32, 27) SC_CASE(96, 48, 27)
  SC_CASE(128, 64, 27)                                     // dec c1 (concat in)
  SC_CASE(32, 16, 1) SC_CASE(64, 32, 1) SC_CASE(96, 48, 1)
  SC_CASE(128, 64, 1)                                      // dec nin
#undef SC_CASE
}

extern "C" void kernel_launch(void* const* d_in, const int* in_sizes, int n_in,
                              void* d_out, int out_size, void* d_ws, size_t ws_size,
                              hipStream_t stream) {
  (void)n_in; (void)out_size; (void)ws_size;
  const int nP[5] = {16, 32, 48, 64, 80};

  // ---- input index map ----
  // Top-level dict in setup_inputs() insertion order:
  //   params, features, nbrs[5], down_idx[4], up_parent[4], up_koff[4]
  // params flattened in JAX leaf order (dict keys sorted, None dropped):
  //   dec(4x16) < enc(4x15 + 12) < in_conv < lin_b < lin_w < out_bn(g,b)
  DecIdx dec[4]; EncIdx enc[5];
  int idx = 0;
  for (int j = 0; j < 4; j++) {
    dec[j].blk[0].bn1g = idx++; dec[j].blk[0].bn1b = idx++;
    dec[j].blk[0].bn2g = idx++; dec[j].blk[0].bn2b = idx++;
    dec[j].blk[0].c1   = idx++; dec[j].blk[0].c2   = idx++;
    dec[j].blk[0].nin  = idx++;
    dec[j].blk[1].bn1g = idx++; dec[j].blk[1].bn1b = idx++;
    dec[j].blk[1].bn2g = idx++; dec[j].blk[1].bn2b = idx++;
    dec[j].blk[1].c1   = idx++; dec[j].blk[1].c2   = idx++;
    dec[j].blk[1].nin  = -1;
    dec[j].bng = idx++; dec[j].bnb = idx++; dec[j].w = idx++;
  }
  for (int i = 0; i < 5; i++) {
    for (int r = 0; r < 2; r++) {
      enc[i].blk[r].bn1g = idx++; enc[i].blk[r].bn1b = idx++;
      enc[i].blk[r].bn2g = idx++; enc[i].blk[r].bn2b = idx++;
      enc[i].blk[r].c1   = idx++; enc[i].blk[r].c2   = idx++;
      enc[i].blk[r].nin  = -1;
    }
    if (i < 4) { enc[i].dbng = idx++; enc[i].dbnb = idx++; enc[i].dw = idx++; }
    else       { enc[i].dbng = enc[i].dbnb = enc[i].dw = -1; }
  }
  const int iInConv = idx++, iLinB = idx++, iLinW = idx++;
  const int iOutG = idx++, iOutB = idx++;
  const int iFeat = idx++;
  int iNbr[5], iDown[4], iUpP[4], iUpK[4];
  for (int l = 0; l < 5; l++) iNbr[l]  = idx++;
  for (int l = 0; l < 4; l++) iDown[l] = idx++;
  for (int l = 0; l < 4; l++) iUpP[l]  = idx++;
  for (int l = 0; l < 4; l++) iUpK[l]  = idx++;
  const int NLEAF = idx;

  auto Fp = [&](int i) { return (const float*)d_in[i]; };
  auto Ip = [&](int i) { return (const int*)d_in[i]; };

  int Nl[5];
  for (int l = 0; l < 5; l++) Nl[l] = in_sizes[iNbr[l]] / 27;
  const int N0 = Nl[0];

  // ---- workspace arena ----
  float* ws = (float*)d_ws;
  size_t off = 0;
  float* stats = ws + off; off += 256;                 // scale+shift (C<=128)
  float* feats[5];
  for (int l = 0; l < 5; l++) { feats[l] = ws + off; off += (size_t)(Nl[l] + 1) * nP[l]; }
  size_t tmpSz = 0;
  for (int l = 0; l < 5; l++) {
    size_t s = (size_t)(Nl[l] + 1) * (size_t)(2 * nP[l]);
    if (s > tmpSz) tmpSz = s;
  }
  tmpSz = (tmpSz + 15) & ~(size_t)15;
  float* P[4];
  for (int p = 0; p < 4; p++) { P[p] = ws + off; off += tmpSz; }
  __bf16* HB = (__bf16*)(ws + off); off += tmpSz / 2 + 16;   // bf16 h buffer
  __bf16* XB = (__bf16*)(ws + off); off += tmpSz / 2 + 16;   // bf16 x (nin/stem)
  __bf16* wArena = (__bf16*)(ws + off);                      // packed weights
  size_t wOff = 0;                                            // in bf16 elems

  // ---- pre-pack all conv weights into B-fragment layout ----
  const __bf16* pkw[200];
  for (int i2 = 0; i2 < NLEAF; i2++) pkw[i2] = nullptr;
  auto packW = [&](int pidx, int C, int D, int K) {
    int CB = (C + 31) / 32;
    size_t elems = (size_t)K * CB * D * 32;
    __bf16* dst = wArena + wOff;
    wOff += (elems + 15) & ~(size_t)15;
    int tot = (int)elems;
    repack_w_kernel<<<(tot + 255) / 256, 256, 0, stream>>>(Fp(pidx), dst, C, D, K, CB);
    pkw[pidx] = dst;
  };
  packW(iInConv, 1, 16, 27);
  for (int i = 0; i < 5; i++) {
    int C = nP[i];
    for (int r = 0; r < 2; r++) {
      packW(enc[i].blk[r].c1, C, C, 27);
      packW(enc[i].blk[r].c2, C, C, 27);
    }
    if (i < 4) packW(enc[i].dw, nP[i], nP[i + 1], 8);
  }
  for (int j = 0; j < 4; j++) {
    int i = 3 - j, B = nP[i], CC = 2 * B;
    packW(dec[j].blk[0].c1, CC, B, 27);
    packW(dec[j].blk[0].c2, B, B, 27);
    packW(dec[j].blk[0].nin, CC, B, 1);
    packW(dec[j].blk[1].c1, B, B, 27);
    packW(dec[j].blk[1].c2, B, B, 27);
  }

  // ---- launch helpers ----
  auto bnrelu = [&](const float* x, int ig, int ib, __bf16* y, int N, int C) {
    bn_stats_kernel<<<C, 256, 0, stream>>>(x, Fp(ig), Fp(ib), stats, N, C);
    int tot = (N + 1) * C;
    bn_apply_kernel<<<(tot + 255) / 256, 256, 0, stream>>>(x, stats, y, N, C);
  };
  auto cvtbf = [&](const float* x, __bf16* y, int N, int C) {
    int tot = (N + 1) * C;
    cvt_bf16_kernel<<<(tot + 255) / 256, 256, 0, stream>>>(x, y, N, C);
  };
  auto subconv = [&](const __bf16* X, const int* nbr, const __bf16* Wp, float* Y,
                     int N, int Nsrc, int C, int D, int K, int ldY, int colOff,
                     int accFlag) {
    SubArgs a{X, nbr, Wp, Y, N, Nsrc, ldY, colOff, accFlag, stream};
    subconv_dispatch(C, D, K, a);
  };
  auto copyc = [&](const float* s, float* d, int N, int W, int lds, int ldd,
                   int co) {
    int tot = N * W;
    copy_cols_kernel<<<(tot + 255) / 256, 256, 0, stream>>>(s, d, N, W, lds, ldd, co);
  };
  auto zero = [&](float* p, int n) {
    zero_kernel<<<(n + 255) / 256, 256, 0, stream>>>(p, n);
  };
  // ResNet block: out = skip(x) + c2(bnrelu(c1(bnrelu(x)))). T = P[3].
  auto run_block = [&](const float* x, float* out, const BlkIdx& b,
                       const int* nbr, int N, int Cin, int Cout) {
    float* T = P[3];
    bnrelu(x, b.bn1g, b.bn1b, HB, N, Cin);                      // HB:[N+1,Cin]
    subconv(HB, nbr, pkw[b.c1], T, N, N, Cin, Cout, 27, Cout, 0, 0);
    bnrelu(T, b.bn2g, b.bn2b, HB, N, Cout);                     // HB:[N+1,Cout]
    if (b.nin >= 0) {
      cvtbf(x, XB, N, Cin);
      subconv(XB, nullptr, pkw[b.nin], out, N, N, Cin, Cout, 1, Cout, 0, 0);
    } else {
      copyc(x, out, N, Cout, Cin, Cout, 0);
    }
    subconv(HB, nbr, pkw[b.c2], out, N, N, Cout, Cout, 27, Cout, 0, 1); // +=skip
  };

  float* cur = P[0];
  float* alt = P[1];

  // ---- stem: in_conv (C=1 -> 16) on bf16-converted, padded features ----
  cvtbf(Fp(iFeat), XB, N0, 1);
  subconv(XB, Ip(iNbr[0]), pkw[iInConv], cur, N0, N0, 1, 16, 27, 16, 0, 0);

  // ---- encoder ----
  for (int i = 0; i < 5; i++) {
    int C = nP[i], N = Nl[i];
    for (int r = 0; r < 2; r++) {
      run_block(cur, alt, enc[i].blk[r], Ip(iNbr[i]), N, C, C);
      float* t = cur; cur = alt; alt = t;
    }
    copyc(cur, feats[i], N, C, C, C, 0);               // save skip features
    if (i < 4) {
      bnrelu(cur, enc[i].dbng, enc[i].dbnb, HB, N, C);     // pad row zeroed
      int Nn = Nl[i + 1], Cn = nP[i + 1];
      subconv(HB, Ip(iDown[i]), pkw[enc[i].dw], alt, Nn, N, C, Cn, 8, Cn, 0, 0);
      float* t = cur; cur = alt; alt = t;
    }
  }

  // ---- decoder ----
  for (int j = 0; j < 4; j++) {
    int i = 3 - j;
    int A = nP[i + 1], B = nP[i], CC = 2 * B;
    int Nc = Nl[i + 1], Nf = Nl[i];
    bnrelu(cur, dec[j].bng, dec[j].bnb, HB, Nc, A);    // HB:[Nc+1,A]
    // concat buffer in `alt`: [Nf(+pad), 2B] = [ feats[i] | upsample ]
    copyc(feats[i], alt, Nf, B, B, CC, 0);
    {
      int tot = Nf * B;
      deconv_kernel<<<(tot + 255) / 256, 256, 0, stream>>>(
          HB, Ip(iUpP[i]), Ip(iUpK[i]), Fp(dec[j].w), alt, Nf, A, B, CC, B);
    }
    zero(alt + (size_t)Nf * CC, CC);                   // pad row
    { float* t = cur; cur = alt; alt = t; }
    run_block(cur, alt, dec[j].blk[0], Ip(iNbr[i]), Nf, CC, B);
    { float* t = cur; cur = alt; alt = t; }
    run_block(cur, alt, dec[j].blk[1], Ip(iNbr[i]), Nf, B, B);
    { float* t = cur; cur = alt; alt = t; }
  }

  // ---- head: out_bn + linear ----
  bnrelu(cur, iOutG, iOutB, HB, N0, 16);
  {
    int tot = N0 * 5;
    head_kernel<<<(tot + 255) / 256, 256, 0, stream>>>(
        HB, Fp(iLinW), Fp(iLinB), (float*)d_out, N0, 16, 5);
  }
}
